// Codec_78323023610349
// MI455X (gfx1250) — compile-verified, matrix operand order checked by reading, not run on hardware
//
#include <hip/hip_runtime.h>
#include <hip/hip_bf16.h>

// ---------------------------------------------------------------------------
// Types for gfx1250 WMMA (wave32): v_wmma_f32_16x16x32_bf16
// ---------------------------------------------------------------------------
typedef __attribute__((ext_vector_type(16))) __bf16 v16bf;
typedef __attribute__((ext_vector_type(8)))  float  v8f;

union Frag16 {
    v16bf  v;
    __bf16 h[16];
    uint4  q[2];
};
static_assert(sizeof(Frag16) == 32, "frag must be 32 bytes");

// Native converts: backend emits v_cvt_pk_bf16_f32 / v_cvt_f32_bf16 class ops
__device__ __forceinline__ __bf16 f2bf(float f) { return (__bf16)f; }
__device__ __forceinline__ float bf2f(unsigned short s) {
    unsigned u = ((unsigned)s) << 16;
    return __builtin_bit_cast(float, u);
}

// tanh from HW transcendentals: 1 - 2/(e^2x + 1); saturates to +/-1 correctly
__device__ __forceinline__ float fast_tanh(float x) {
    const float t = __expf(2.0f * x);
    return 1.0f - 2.0f * __builtin_amdgcn_rcpf(t + 1.0f);
}

__device__ __forceinline__ void store_out(float* p, float v) { *p = v; }
__device__ __forceinline__ void store_out(__bf16* p, float v) { *p = f2bf(v); }

// ---------------------------------------------------------------------------
// Tiled WMMA GEMM:  out[M,N] = act( A[M,K] @ W[0:K, N]  + W[K, N] )
//   W is (K+1) x N row-major f32, last row = bias.
//   A is row-major, either f32 (converted inline) or bf16.
// Tile: 128(M) x 128(N) per 256-thread workgroup; 8 waves x (16x128); BK = 32.
// Per K-step each wave: 1 A-fragment (global), 8 B-fragments (LDS, double-
// buffered so ds_load_b128 of tile t+1 pipelines under WMMA t), 8 WMMAs.
// ---------------------------------------------------------------------------
template <bool AF32, bool TANH, typename OUT>
__global__ __launch_bounds__(256) void wmma_gemm(const void* __restrict__ Ap,
                                                 const float* __restrict__ W,
                                                 OUT* __restrict__ out,
                                                 int M, int N, int K) {
    constexpr int BK = 32;
    __shared__ __bf16 Bs[128][40];          // [n][k] bf16, +8 pad: rows 80B apart (16B aligned)

    const int tid   = threadIdx.x;
    const int wave  = tid >> 5;
    const int lane  = tid & 31;
    const int half  = lane >> 4;            // K-half select per ISA A/B 16-bit layout
    const int l15   = lane & 15;

    const int nblock  = blockIdx.x * 128;
    const int rowbase = blockIdx.y * 128 + wave * 16;
    const int row_g   = rowbase + l15;      // A-matrix row for this lane (both halves)

    v8f acc[8];
#pragma unroll
    for (int t = 0; t < 8; ++t) acc[t] = v8f{0.f, 0.f, 0.f, 0.f, 0.f, 0.f, 0.f, 0.f};

    // B-fragment fetch: lane holds col = t*16 + l15, two aligned 8-elem K runs
    auto loadB = [&](Frag16& f, int t) {
        const int col = t * 16 + l15;
        f.q[0] = *(const uint4*)&Bs[col][8 * half];
        f.q[1] = *(const uint4*)&Bs[col][16 + 8 * half];
    };

    for (int kb = 0; kb < K; kb += BK) {
        // ---- stage B tile (BK x 128 f32) -> LDS transposed bf16 ------------
#pragma unroll
        for (int i = 0; i < 4; ++i) {
            const int idx4 = tid + 256 * i;         // 0..1023 float4s (32x128 f32)
            const int kk   = idx4 >> 5;             // 0..31
            const int n4   = (idx4 & 31) << 2;      // 0..124
            const float4 wv = *(const float4*)&W[(size_t)(kb + kk) * N + nblock + n4];
            Bs[n4 + 0][kk] = f2bf(wv.x);
            Bs[n4 + 1][kk] = f2bf(wv.y);
            Bs[n4 + 2][kk] = f2bf(wv.z);
            Bs[n4 + 3][kk] = f2bf(wv.w);
            if (kb + BK < K)  // global_prefetch_b8 for next weight slab
                __builtin_prefetch(&W[(size_t)(kb + BK + kk) * N + nblock + n4], 0, 3);
        }
        __syncthreads();

        // ---- A fragment: 16x32, lane holds row=l15, two 8-elem K runs ------
        // e in [0,8): K = kb + 8*half + e ; e in [8,16): K = kb + 16 + 8*half + (e-8)
        Frag16 af;
        if constexpr (AF32) {
            const float* pa = (const float*)Ap + (size_t)row_g * K + kb;
            const float4 f0 = *(const float4*)(pa + 8 * half);
            const float4 f1 = *(const float4*)(pa + 8 * half + 4);
            const float4 f2 = *(const float4*)(pa + 16 + 8 * half);
            const float4 f3 = *(const float4*)(pa + 16 + 8 * half + 4);
            af.h[0]  = f2bf(f0.x); af.h[1]  = f2bf(f0.y);
            af.h[2]  = f2bf(f0.z); af.h[3]  = f2bf(f0.w);
            af.h[4]  = f2bf(f1.x); af.h[5]  = f2bf(f1.y);
            af.h[6]  = f2bf(f1.z); af.h[7]  = f2bf(f1.w);
            af.h[8]  = f2bf(f2.x); af.h[9]  = f2bf(f2.y);
            af.h[10] = f2bf(f2.z); af.h[11] = f2bf(f2.w);
            af.h[12] = f2bf(f3.x); af.h[13] = f2bf(f3.y);
            af.h[14] = f2bf(f3.z); af.h[15] = f2bf(f3.w);
        } else {
            const __bf16* pa = (const __bf16*)Ap + (size_t)row_g * K + kb;
            af.q[0] = *(const uint4*)(pa + 8 * half);
            af.q[1] = *(const uint4*)(pa + 16 + 8 * half);
        }

        // ---- 8 WMMA tiles along N, B-fragment double buffered --------------
        Frag16 bfa, bfb;
        loadB(bfa, 0);
#pragma unroll
        for (int t = 0; t < 8; ++t) {
            Frag16& cur = (t & 1) ? bfb : bfa;
            Frag16& nxt = (t & 1) ? bfa : bfb;
            if (t < 7) loadB(nxt, t + 1);
            acc[t] = __builtin_amdgcn_wmma_f32_16x16x32_bf16(
                false, af.v, false, cur.v, (short)0, acc[t], false, false);
        }
        __syncthreads();
    }

    // ---- epilogue: bias (+ tanh), D layout row = r + 8*half, col = l15 -----
#pragma unroll
    for (int t = 0; t < 8; ++t) {
        const int colg = nblock + t * 16 + l15;
        const float bias = W[(size_t)K * N + colg];
#pragma unroll
        for (int r = 0; r < 8; ++r) {
            float v = acc[t][r] + bias;
            if (TANH) v = fast_tanh(v);
            store_out(&out[(size_t)(rowbase + r + 8 * half) * N + colg], v);
        }
    }
}

// ---------------------------------------------------------------------------
// VQ: per (row b, chunk c): d_k = ||x - C_k||, r = softmax(-d/8),
//     idx = argmax(r - noise_level*U), y = C_idx  (STE forward value)
// One wave32 per task; 4 waves / block; codebook staged bf16 in LDS.
// ---------------------------------------------------------------------------
__device__ __forceinline__ float hash_uniform(unsigned a, unsigned b) {
    unsigned h = a * 0x9E3779B1u ^ (b * 0x85EBCA77u + 0xC2B2AE3Du);
    h ^= h >> 16; h *= 0x7FEB352Du;
    h ^= h >> 15; h *= 0x846CA68Bu;
    h ^= h >> 16;
    return (float)(h >> 8) * (1.0f / 16777216.0f);
}

__global__ __launch_bounds__(128) void vq_kernel(const float* __restrict__ h2,
                                                 const float* __restrict__ cb,
                                                 const int* __restrict__ nl_p,
                                                 __bf16* __restrict__ y) {
    __shared__ __bf16 cbs[256][66];   // pad 66 -> row stride 33 dwords: conflict-free
    __shared__ float  xs[4][64];

    const int tid  = threadIdx.x;
    const int wave = tid >> 5;
    const int lane = tid & 31;

    // stage codebook (256x64 f32 -> bf16 LDS), 128 threads x 32 float4
#pragma unroll 8
    for (int i = 0; i < 32; ++i) {
        const int idx4 = i * 128 + tid;      // 0..4095
        const int k    = idx4 >> 4;
        const int j4   = (idx4 & 15) << 2;
        const float4 c = *(const float4*)&cb[(size_t)k * 64 + j4];
        cbs[k][j4 + 0] = f2bf(c.x);
        cbs[k][j4 + 1] = f2bf(c.y);
        cbs[k][j4 + 2] = f2bf(c.z);
        cbs[k][j4 + 3] = f2bf(c.w);
    }

    const int task = blockIdx.x * 4 + wave;
    const int b    = task >> 2;
    const int ch   = task & 3;
    const float2 xv = *(const float2*)(h2 + (size_t)b * 256 + ch * 64 + 2 * lane);
    xs[wave][2 * lane]     = xv.x;
    xs[wave][2 * lane + 1] = xv.y;
    __syncthreads();

    float x2 = 0.f;
    for (int j = 0; j < 64; ++j) { const float xj = xs[wave][j]; x2 += xj * xj; }

    const float nl = (float)(*nl_p);

    // per-lane 8 codes: k = kk*32 + lane
    float sc[8];
#pragma unroll
    for (int kk = 0; kk < 8; ++kk) {
        const int k = kk * 32 + lane;
        float dot = 0.f, c2 = 0.f;
        for (int j = 0; j < 64; j += 2) {
            const unsigned pair = *(const unsigned*)&cbs[k][j];
            const float c0 = bf2f((unsigned short)(pair & 0xFFFFu));
            const float c1 = bf2f((unsigned short)(pair >> 16));
            dot += xs[wave][j] * c0 + xs[wave][j + 1] * c1;
            c2  += c0 * c0 + c1 * c1;
        }
        const float d = sqrtf(fmaxf(x2 - 2.f * dot + c2, 0.f));
        sc[kk] = -d * 0.125f;                // / sqrt(64)
    }

    // softmax over 256 (wave32 butterflies)
    float m = sc[0];
#pragma unroll
    for (int kk = 1; kk < 8; ++kk) m = fmaxf(m, sc[kk]);
#pragma unroll
    for (int off = 16; off >= 1; off >>= 1) m = fmaxf(m, __shfl_xor(m, off, 32));

    float e[8];
    float sum = 0.f;
#pragma unroll
    for (int kk = 0; kk < 8; ++kk) { e[kk] = __expf(sc[kk] - m); sum += e[kk]; }
#pragma unroll
    for (int off = 16; off >= 1; off >>= 1) sum += __shfl_xor(sum, off, 32);
    const float inv = 1.f / sum;

    // argmax(r - nl*noise), first-index tie-break
    float bestv = -3.4e38f;
    int   besti = 0;
#pragma unroll
    for (int kk = 0; kk < 8; ++kk) {
        const int k = kk * 32 + lane;
        const float score = e[kk] * inv - nl * hash_uniform((unsigned)task, (unsigned)k);
        if (score > bestv || (score == bestv && k < besti)) { bestv = score; besti = k; }
    }
#pragma unroll
    for (int off = 16; off >= 1; off >>= 1) {
        const float ov = __shfl_xor(bestv, off, 32);
        const int   oi = __shfl_xor(besti, off, 32);
        if (ov > bestv || (ov == bestv && oi < besti)) { bestv = ov; besti = oi; }
    }

    // y = codebook[besti] (full-precision source, stored bf16 for decoder GEMM)
    const float2 cw = *(const float2*)&cb[(size_t)besti * 64 + 2 * lane];
    __bf16* yr = y + (size_t)b * 256 + ch * 64 + 2 * lane;
    yr[0] = f2bf(cw.x);
    yr[1] = f2bf(cw.y);
}

// ---------------------------------------------------------------------------
// Launch: enc1 -> enc2 -> vq -> dec1 -> dec2
// ---------------------------------------------------------------------------
extern "C" void kernel_launch(void* const* d_in, const int* in_sizes, int n_in,
                              void* d_out, int out_size, void* d_ws, size_t ws_size,
                              hipStream_t stream) {
    const float* x    = (const float*)d_in[0];   // [B, 3072]
    const float* wb1e = (const float*)d_in[1];   // [3073, 16384]
    const float* wb2e = (const float*)d_in[2];   // [16385, 256]
    const float* wb1d = (const float*)d_in[3];   // [257, 16384]
    const float* wb2d = (const float*)d_in[4];   // [16385, 3072]
    const float* cb1  = (const float*)d_in[5];   // [256, 64] (used for all 4 chunks)
    const int*   nl   = (const int*)d_in[9];
    float* out = (float*)d_out;

    const int B = in_sizes[0] / 3072;            // 4096

    char* ws = (char*)d_ws;
    __bf16* h_buf = (__bf16*)ws;                                        // [B,16384] bf16 (h1 then h3)
    float*  hb2   = (float*)(ws + (size_t)B * 16384 * 2);               // [B,256] f32
    __bf16* ybuf  = (__bf16*)(ws + (size_t)B * 16384 * 2 +
                                   (size_t)B * 256 * 4);                // [B,256] bf16

    const dim3 blk(256);
    // h1 = tanh(x @ W1e + b1e)            M=B, N=16384, K=3072
    wmma_gemm<true,  true,  __bf16><<<dim3(16384 / 128, B / 128), blk, 0, stream>>>(
        x, wb1e, h_buf, B, 16384, 3072);
    // h2 = tanh(h1 @ W2e + b2e)           M=B, N=256, K=16384
    wmma_gemm<false, true,  float><<<dim3(256 / 128, B / 128), blk, 0, stream>>>(
        h_buf, wb2e, hb2, B, 256, 16384);
    // vector quantize 4 chunks against codebook1
    vq_kernel<<<B, 128, 0, stream>>>(hb2, cb1, nl, ybuf);
    // h3 = tanh(y @ W1d + b1d)            M=B, N=16384, K=256
    wmma_gemm<false, true,  __bf16><<<dim3(16384 / 128, B / 128), blk, 0, stream>>>(
        ybuf, wb1d, h_buf, B, 16384, 256);
    // out = h3 @ W2d + b2d                M=B, N=3072, K=16384
    wmma_gemm<false, false, float><<<dim3(3072 / 128, B / 128), blk, 0, stream>>>(
        h_buf, wb2d, out, B, 3072, 16384);
}
